// DeepGNN_84112639524973
// MI455X (gfx1250) — compile-verified, matrix operand order checked by reading, not run on hardware
//
#include <hip/hip_runtime.h>
#include <hip/hip_bf16.h>
#include <stdint.h>

// ---------------------------------------------------------------------------
// CDNA5 (gfx1250) wave32 WMMA types
// ---------------------------------------------------------------------------
typedef __attribute__((ext_vector_type(16))) __bf16 v16bf;
typedef __attribute__((ext_vector_type(8)))  float  v8f;

__device__ __forceinline__ unsigned short f32_to_bf16(float f) {
    unsigned u = __float_as_uint(f);
    unsigned r = u + 0x7FFFu + ((u >> 16) & 1u);   // round-to-nearest-even
    return (unsigned short)(r >> 16);
}
__device__ __forceinline__ float bf16_to_f32(unsigned short s) {
    return __uint_as_float(((unsigned)s) << 16);
}
__device__ __forceinline__ float sigmoidf(float x) {
    return 1.0f / (1.0f + __expf(-x));
}

// ---------------------------------------------------------------------------
// Utility kernels
// ---------------------------------------------------------------------------
__global__ void fill_u32_kernel(uint32_t* p, uint32_t v, long n) {
    long i = (long)blockIdx.x * blockDim.x + threadIdx.x;
    if (i < n) p[i] = v;
}

__global__ void cvt_f32_bf16_kernel(const float* __restrict__ in,
                                    unsigned short* __restrict__ out, long n) {
    long i = (long)blockIdx.x * blockDim.x + threadIdx.x;
    if (i < n) out[i] = f32_to_bf16(in[i]);
}

// W is [128,128] row-major (K x N); write Bt[n*128 + k] = bf16(W[k*128 + n])
__global__ void cvt_transpose128_kernel(const float* __restrict__ W,
                                        unsigned short* __restrict__ Wt) {
    int idx = blockIdx.x * blockDim.x + threadIdx.x;   // 16384 total
    if (idx >= 128 * 128) return;
    int k = idx >> 7, n = idx & 127;
    Wt[n * 128 + k] = f32_to_bf16(W[idx]);
}

// ---------------------------------------------------------------------------
// Degree / normalization
// ---------------------------------------------------------------------------
__global__ void deg_count_kernel(const int* __restrict__ dst, float* deg, int E) {
    int e = blockIdx.x * blockDim.x + threadIdx.x;
    if (e < E) atomicAdd(&deg[dst[e]], 1.0f);
}
__global__ void deg_finish_kernel(float* dis, int N) {
    int i = blockIdx.x * blockDim.x + threadIdx.x;
    if (i < N) dis[i] = rsqrtf(dis[i] + 1.0f);
}

// ---------------------------------------------------------------------------
// WMMA GEMM: C[M, NCOLS](f32) = A[M,128](bf16, row-major) @ B(bf16, Bt layout:
// Bt[n*128 + k]).  8 waves/block, one 16x16 tile per wave.
// NCOLS is a compile-time constant so C-row strides become immediate offsets.
// ---------------------------------------------------------------------------
template <int NCOLS>
__global__ __launch_bounds__(256) void wmma_gemm_bf16_kernel(
    const unsigned short* __restrict__ A, const unsigned short* __restrict__ Bt,
    float* __restrict__ C, int M) {
    int lane = threadIdx.x & 31;
    int wave = threadIdx.x >> 5;
    int tn   = blockIdx.y * 8 + wave;
    if (tn * 16 >= NCOLS) return;              // wave-uniform exit
    int tm   = blockIdx.x;
    int l16  = lane & 15;
    int half = lane >> 4;

    int arow = tm * 16 + l16;
    if (arow >= M) arow = M - 1;               // clamped load; store guarded
    int bcol = tn * 16 + l16;

    v8f acc;
#pragma unroll
    for (int r = 0; r < 8; ++r) acc[r] = 0.0f;

    const unsigned short* arp = A  + (size_t)arow * 128;
    const unsigned short* brp = Bt + (size_t)bcol * 128;
#pragma unroll
    for (int k0 = 0; k0 < 128; k0 += 32) {
        int k = k0 + half * 16;
        v16bf a = *(const v16bf*)(arp + k);
        v16bf b = *(const v16bf*)(brp + k);
        acc = __builtin_amdgcn_wmma_f32_16x16x32_bf16(false, a, false, b,
                                                      (short)0, acc, false, false);
    }

    int crow0 = tm * 16 + half * 8;
    float* cp = C + (size_t)crow0 * NCOLS + tn * 16 + l16;
    if (tm * 16 + 16 <= M) {                   // wave-uniform: full tile
#pragma unroll
        for (int r = 0; r < 8; ++r) cp[(size_t)r * NCOLS] = acc[r];
    } else {
#pragma unroll
        for (int r = 0; r < 8; ++r)
            if (crow0 + r < M) cp[(size_t)r * NCOLS] = acc[r];
    }
}

// ---------------------------------------------------------------------------
// GCN scatter: agg[dst] += dis[src]*dis[dst] * xw[src]   (per-edge, float4)
// ---------------------------------------------------------------------------
__global__ void gcn_scatter_kernel(const int* __restrict__ src,
                                   const int* __restrict__ dst,
                                   const float* __restrict__ dis,
                                   const float* __restrict__ xw,
                                   float* __restrict__ agg, long total) {
    long i = (long)blockIdx.x * blockDim.x + threadIdx.x;   // total = E*32
    if (i >= total) return;
    int e  = (int)(i >> 5);
    int f4 = (int)(i & 31);
    int s = src[e], d = dst[e];
    float w = dis[s] * dis[d];
    float4 v = *(const float4*)(xw + (size_t)s * 128 + f4 * 4);
    float* base = agg + (size_t)d * 128 + f4 * 4;
    atomicAdd(base + 0, w * v.x);
    atomicAdd(base + 1, w * v.y);
    atomicAdd(base + 2, w * v.z);
    atomicAdd(base + 3, w * v.w);
}

// out = relu(agg + xw * dis^2 + b)  -> bf16 activation buffer
__global__ void gcn_combine_kernel(const float* __restrict__ agg,
                                   const float* __restrict__ xw,
                                   const float* __restrict__ dis,
                                   const float* __restrict__ b,
                                   unsigned short* __restrict__ outbf, long n) {
    long i = (long)blockIdx.x * blockDim.x + threadIdx.x;   // n = N*128
    if (i >= n) return;
    int node = (int)(i >> 7);
    int f    = (int)(i & 127);
    float ds = dis[node];
    float v  = agg[i] + xw[i] * ds * ds + b[f];
    outbf[i] = f32_to_bf16(fmaxf(v, 0.0f));
}

// ---------------------------------------------------------------------------
// LSTM gates: gates[G,512] = xt@W_ih^T + h@W_hh^T + (b_ih + b_hh)
// xt row g is seqbf[(g*L + t)*128 ..]. W_ih/W_hh are [512,128] row-major,
// which is exactly the Bt layout for B = W^T.  ldc = 512 (constant).
// ---------------------------------------------------------------------------
__global__ __launch_bounds__(256) void lstm_gates_kernel(
    const unsigned short* __restrict__ seqbf, int t, int L,
    const unsigned short* __restrict__ hbf,
    const unsigned short* __restrict__ Wihb,
    const unsigned short* __restrict__ Whhb,
    const float* __restrict__ b_ih, const float* __restrict__ b_hh,
    float* __restrict__ gates, int G) {
    int lane = threadIdx.x & 31;
    int wave = threadIdx.x >> 5;
    int tm   = blockIdx.x;                    // row tile over G (padded)
    int tn   = blockIdx.y * 8 + wave;         // col tile over 512
    int l16  = lane & 15;
    int half = lane >> 4;

    int g  = tm * 16 + l16;
    int gc = (g < G) ? g : (G - 1);
    int bcol = tn * 16 + l16;

    float bias = b_ih[bcol] + b_hh[bcol];
    v8f acc;
#pragma unroll
    for (int r = 0; r < 8; ++r) acc[r] = bias;

    const unsigned short* xrow = seqbf + ((size_t)gc * L + t) * 128;
    const unsigned short* hrow = hbf + (size_t)gc * 128;
    const unsigned short* bih  = Wihb + (size_t)bcol * 128;
    const unsigned short* bhh  = Whhb + (size_t)bcol * 128;
#pragma unroll
    for (int k0 = 0; k0 < 128; k0 += 32) {
        int k = k0 + half * 16;
        v16bf ax = *(const v16bf*)(xrow + k);
        v16bf bx = *(const v16bf*)(bih + k);
        acc = __builtin_amdgcn_wmma_f32_16x16x32_bf16(false, ax, false, bx,
                                                      (short)0, acc, false, false);
        v16bf ah = *(const v16bf*)(hrow + k);
        v16bf bh = *(const v16bf*)(bhh + k);
        acc = __builtin_amdgcn_wmma_f32_16x16x32_bf16(false, ah, false, bh,
                                                      (short)0, acc, false, false);
    }

    int crow0 = tm * 16 + half * 8;
    float* cp = gates + (size_t)crow0 * 512 + tn * 16 + l16;
    if (tm * 16 + 16 <= G) {                  // wave-uniform: full tile
#pragma unroll
        for (int r = 0; r < 8; ++r) cp[(size_t)r * 512] = acc[r];
    } else {
#pragma unroll
        for (int r = 0; r < 8; ++r)
            if (crow0 + r < G) cp[(size_t)r * 512] = acc[r];
    }
}

__global__ void lstm_cell_kernel(const float* __restrict__ gates,
                                 float* __restrict__ cs,
                                 unsigned short* __restrict__ hbf, int G) {
    int i = blockIdx.x * blockDim.x + threadIdx.x;   // G*128
    if (i >= G * 128) return;
    int g = i >> 7, j = i & 127;
    const float* gr = gates + (size_t)g * 512;
    float ig = sigmoidf(gr[j]);
    float fg = sigmoidf(gr[128 + j]);
    float gg = tanhf(gr[256 + j]);
    float og = sigmoidf(gr[384 + j]);
    float c  = fg * cs[i] + ig * gg;
    cs[i]  = c;
    hbf[i] = f32_to_bf16(og * tanhf(c));
}

// out[G,C] = h @ lin_W + lin_b
__global__ void final_linear_kernel(const unsigned short* __restrict__ hbf,
                                    const float* __restrict__ W,
                                    const float* __restrict__ b,
                                    float* __restrict__ out, int G, int C) {
    int i = blockIdx.x * blockDim.x + threadIdx.x;
    if (i >= G * C) return;
    int g = i / C, c = i % C;
    float s = b[c];
    const unsigned short* hr = hbf + (size_t)g * 128;
#pragma unroll 4
    for (int k = 0; k < 128; ++k) s += bf16_to_f32(hr[k]) * W[k * C + c];
    out[i] = s;
}

// ---------------------------------------------------------------------------
// Host launch
// ---------------------------------------------------------------------------
static inline dim3 blocks1d(long n) { return dim3((unsigned)((n + 255) / 256)); }

extern "C" void kernel_launch(void* const* d_in, const int* in_sizes, int n_in,
                              void* d_out, int out_size, void* d_ws, size_t ws_size,
                              hipStream_t stream) {
    const float* x      = (const float*)d_in[0];
    const int*   ei     = (const int*)d_in[1];     // [2, E]
    const float* W1     = (const float*)d_in[3];
    const float* b1     = (const float*)d_in[4];
    const float* W2     = (const float*)d_in[5];
    const float* b2     = (const float*)d_in[6];
    const float* W3     = (const float*)d_in[7];
    const float* b3     = (const float*)d_in[8];
    const float* W4     = (const float*)d_in[9];
    const float* b4     = (const float*)d_in[10];
    const float* W_ih   = (const float*)d_in[11];  // [512,128]
    const float* W_hh   = (const float*)d_in[12];  // [512,128]
    const float* b_ih   = (const float*)d_in[13];
    const float* b_hh   = (const float*)d_in[14];
    const float* lin_W  = (const float*)d_in[15];  // [128,10]
    const float* lin_b  = (const float*)d_in[16];
    float*       out    = (float*)d_out;

    const int H = 128;
    const int N = in_sizes[0] / H;            // 50000
    const int E = in_sizes[1] / 2;            // 800000
    const int C = in_sizes[16];               // 10
    const int G = out_size / C;               // 500
    const int L = N / G;                      // 100

    // workspace partition (256B aligned)
    uint8_t* p = (uint8_t*)d_ws;
    auto take = [&](size_t bytes) -> void* {
        void* r = (void*)p;
        p += (bytes + 255) & ~(size_t)255;
        return r;
    };
    float*          dis   = (float*)take((size_t)N * 4);
    float*          xw    = (float*)take((size_t)N * H * 4);
    float*          agg   = (float*)take((size_t)N * H * 4);
    unsigned short* xbf   = (unsigned short*)take((size_t)N * H * 2);
    unsigned short* hAbf  = (unsigned short*)take((size_t)N * H * 2);
    unsigned short* hBbf  = (unsigned short*)take((size_t)N * H * 2);
    unsigned short* Wt1   = (unsigned short*)take((size_t)H * H * 2);
    unsigned short* Wt2   = (unsigned short*)take((size_t)H * H * 2);
    unsigned short* Wt3   = (unsigned short*)take((size_t)H * H * 2);
    unsigned short* Wt4   = (unsigned short*)take((size_t)H * H * 2);
    unsigned short* Wihb  = (unsigned short*)take((size_t)4 * H * H * 2);
    unsigned short* Whhb  = (unsigned short*)take((size_t)4 * H * H * 2);
    float*          gates = (float*)take((size_t)512 * 512 * 4);
    float*          cs    = (float*)take((size_t)512 * H * 4);
    unsigned short* hsbf  = (unsigned short*)take((size_t)512 * H * 2);

    const int* srcE = ei;
    const int* dstE = ei + E;

    // --- weight prep (bf16) ---
    cvt_transpose128_kernel<<<blocks1d(H * H), 256, 0, stream>>>(W1, Wt1);
    cvt_transpose128_kernel<<<blocks1d(H * H), 256, 0, stream>>>(W2, Wt2);
    cvt_transpose128_kernel<<<blocks1d(H * H), 256, 0, stream>>>(W3, Wt3);
    cvt_transpose128_kernel<<<blocks1d(H * H), 256, 0, stream>>>(W4, Wt4);
    cvt_f32_bf16_kernel<<<blocks1d(4L * H * H), 256, 0, stream>>>(W_ih, Wihb, 4L * H * H);
    cvt_f32_bf16_kernel<<<blocks1d(4L * H * H), 256, 0, stream>>>(W_hh, Whhb, 4L * H * H);
    cvt_f32_bf16_kernel<<<blocks1d((long)N * H), 256, 0, stream>>>(x, xbf, (long)N * H);

    // --- degree / normalization ---
    fill_u32_kernel<<<blocks1d(N), 256, 0, stream>>>((uint32_t*)dis, 0u, N);
    deg_count_kernel<<<blocks1d(E), 256, 0, stream>>>(dstE, dis, E);
    deg_finish_kernel<<<blocks1d(N), 256, 0, stream>>>(dis, N);

    // --- 4 GCN layers (ping-pong bf16 activations) ---
    const unsigned short* Wts[4] = {Wt1, Wt2, Wt3, Wt4};
    const float*          bs[4]  = {b1, b2, b3, b4};
    const unsigned short* ins[4] = {xbf, hAbf, hBbf, hAbf};
    unsigned short*       outs[4]= {hAbf, hBbf, hAbf, hBbf};

    const long nh    = (long)N * H;
    const long escat = (long)E * 32;
    dim3 gemmGrid((unsigned)((N + 15) / 16), 1);

    for (int l = 0; l < 4; ++l) {
        fill_u32_kernel<<<blocks1d(nh), 256, 0, stream>>>((uint32_t*)agg, 0u, nh);
        wmma_gemm_bf16_kernel<128><<<gemmGrid, 256, 0, stream>>>(ins[l], Wts[l], xw, N);
        gcn_scatter_kernel<<<blocks1d(escat), 256, 0, stream>>>(srcE, dstE, dis, xw, agg, escat);
        gcn_combine_kernel<<<blocks1d(nh), 256, 0, stream>>>(agg, xw, dis, bs[l], outs[l], nh);
    }
    const unsigned short* seqbf = hBbf;   // final GCN output = LSTM input sequence

    // --- LSTM over L steps ---
    fill_u32_kernel<<<blocks1d(512L * H), 256, 0, stream>>>((uint32_t*)cs, 0u, 512L * H);
    fill_u32_kernel<<<blocks1d(512L * H / 2), 256, 0, stream>>>((uint32_t*)hsbf, 0u, 512L * H / 2);

    dim3 gatesGrid((unsigned)((G + 15) / 16), 4);   // 512 cols / (16*8 waves)
    for (int t = 0; t < L; ++t) {
        lstm_gates_kernel<<<gatesGrid, 256, 0, stream>>>(seqbf, t, L, hsbf, Wihb, Whhb,
                                                         b_ih, b_hh, gates, G);
        lstm_cell_kernel<<<blocks1d((long)G * H), 256, 0, stream>>>(gates, cs, hsbf, G);
    }

    // --- final linear ---
    final_linear_kernel<<<blocks1d((long)G * C), 256, 0, stream>>>(hsbf, lin_W, lin_b,
                                                                   out, G, C);
}
